// MultiHeadSelfAttention_30837865185679
// MI455X (gfx1250) — matrix operand order repaired and checked by reading, not compiled, not run on hardware
//
#include <hip/hip_runtime.h>
#include <hip/hip_bf16.h>

// ---------------------------------------------------------------------------
// Multi-head self attention for MI455X (gfx1250, wave32, WMMA).
// f32 -> one-pass f16 convert -> async-LDS double-buffered f16 WMMA GEMMs
// (fp32 accumulate) -> flash attention -> f32 output.
// ---------------------------------------------------------------------------

typedef __attribute__((ext_vector_type(16))) _Float16 v16h;
typedef __attribute__((ext_vector_type(8)))  float    v8f;

#define BATCH 2
#define SEQ   2048
#define EMB   1024
#define HEADS 16
#define HDIM  64
#define ROWS  (BATCH * SEQ)        // 4096
#define QKVC  (3 * EMB)            // 3072

static __device__ __forceinline__ v8f wmma_f16(v16h a, v16h b, v8f c) {
  // (neg_a, A, neg_b, B, c_mod, C, reuse_a, reuse_b)
  return __builtin_amdgcn_wmma_f32_16x16x32_f16(false, a, false, b, (short)0, c,
                                                false, false);
}

// ---- CDNA5 async global->LDS copy (ASYNCcnt path, ISA ch.10.7/15.18) -------
// Per-lane 16B copy; low 32 bits of a generic LDS pointer are the LDS offset
static __device__ __forceinline__ void async_copy_b128(_Float16* lds_dst,
                                                       const _Float16* gsrc) {
  asm volatile("global_load_async_to_lds_b128 %0, %1, off"
               :
               : "v"((unsigned)(unsigned long long)lds_dst),
                 "v"((unsigned long long)gsrc)
               : "memory");
}
static __device__ __forceinline__ void wait_async() {
  asm volatile("s_wait_asynccnt 0" ::: "memory");
}

// ---- Fragment loaders (CDNA5 ISA 7.12.2 layouts, wave32) -------------------
// A fragment 16x32 f16: half h -> k = 16*(h>=8) + 8*(lane>=16) + (h&7).
static __device__ __forceinline__ v16h
load_frag_a(const _Float16* lds, int row0, int col0, int stride, int lane) {
  const _Float16* p = lds + (size_t)(row0 + (lane & 15)) * stride + col0;
  const int kb = (lane >> 4) << 3;
  v16h a;
#pragma unroll
  for (int h = 0; h < 16; ++h) a[h] = p[((h >> 3) << 4) + kb + (h & 7)];
  return a;
}

// Same but f32 LDS source (convert on the fly) — used for P fragments.
static __device__ __forceinline__ v16h
load_frag_a_f32(const float* lds, int row0, int col0, int stride, int lane) {
  const float* p = lds + (size_t)(row0 + (lane & 15)) * stride + col0;
  const int kb = (lane >> 4) << 3;
  v16h a;
#pragma unroll
  for (int h = 0; h < 16; ++h)
    a[h] = (_Float16)p[((h >> 3) << 4) + kb + (h & 7)];
  return a;
}

// B fragment 32x16 f16: n = lane&15, half h -> k = 16*(h>=8) + 8*(lane>=16) + (h&7).
static __device__ __forceinline__ v16h
load_frag_b(const _Float16* lds, int k0, int n0, int stride, int lane) {
  const _Float16* p = lds + (size_t)k0 * stride + n0 + (lane & 15);
  const int kb = (lane >> 4) << 3;
  v16h b;
#pragma unroll
  for (int h = 0; h < 16; ++h) b[h] = p[(size_t)(((h >> 3) << 4) + kb + (h & 7)) * stride];
  return b;
}

// ---------------------------------------------------------------------------
// Kernel 0: flat f32 -> f16 conversion (one pass; everything stays in L2).
// ---------------------------------------------------------------------------
__global__ void __launch_bounds__(256)
convert_f16_kernel(const float* __restrict__ in, _Float16* __restrict__ out,
                   int n) {
  const int i = (blockIdx.x * 256 + threadIdx.x) * 8;
  if (i < n) {
#pragma unroll
    for (int j = 0; j < 8; ++j) out[i + j] = (_Float16)in[i + j];
  }
}

// ---------------------------------------------------------------------------
// Kernel 1: QKV = xh @ wqkvh + b_qkv. Q,V -> [B,H,N,D]; K -> [B,H,D,N] (pre-
// transposed so attention B-fragments are contiguous LDS reads).
// ---------------------------------------------------------------------------
#define G_AS 40    // 32 + 8 pad halves (80B rows, 16B multiple)
#define G_BS 136   // 128 + 8 pad halves (272B rows, 16B multiple)

__global__ void __launch_bounds__(256)
qkv_gemm_kernel(const _Float16* __restrict__ xh, const _Float16* __restrict__ w,
                const float* __restrict__ bias, _Float16* __restrict__ q_ws,
                _Float16* __restrict__ kT_ws, _Float16* __restrict__ v_ws) {
  __shared__ __attribute__((aligned(16))) _Float16 As[2][128 * G_AS];
  __shared__ __attribute__((aligned(16))) _Float16 Bs[2][32 * G_BS];
  const int t = threadIdx.x;
  const int wave = t >> 5, lane = t & 31;
  const int row0 = blockIdx.y * 128;
  const int col0 = blockIdx.x * 128;

  v8f acc[8];
#pragma unroll
  for (int i = 0; i < 8; ++i) acc[i] = (v8f){};

  // async-stage one k-tile (A: 128x32 halves, B: 32x128 halves)
  auto stage = [&](int buf, int kk) {
#pragma unroll
    for (int i = 0; i < 2; ++i) {
      int ch = i * 256 + t;                  // A: 4 chunks/row
      int r = ch >> 2, c8 = (ch & 3) << 3;
      async_copy_b128(&As[buf][r * G_AS + c8],
                      xh + (size_t)(row0 + r) * EMB + kk + c8);
    }
#pragma unroll
    for (int i = 0; i < 2; ++i) {
      int ch = i * 256 + t;                  // B: 16 chunks/row
      int r = ch >> 4, c8 = (ch & 15) << 3;
      async_copy_b128(&Bs[buf][r * G_BS + c8],
                      w + (size_t)(kk + r) * QKVC + col0 + c8);
    }
  };

  stage(0, 0);
  wait_async();
  __syncthreads();

  for (int kk = 0; kk < EMB; kk += 32) {
    const int cur = (kk >> 5) & 1;
    if (kk + 32 < EMB) stage(cur ^ 1, kk + 32);   // overlap DMA with WMMA
    v16h a = load_frag_a(As[cur], 16 * wave, 0, G_AS, lane);
#pragma unroll
    for (int ct = 0; ct < 8; ++ct) {
      v16h b = load_frag_b(Bs[cur], 0, 16 * ct, G_BS, lane);
      acc[ct] = wmma_f16(a, b, acc[ct]);
    }
    wait_async();
    __syncthreads();
  }

  const int mo = (lane >> 4) << 3;           // C/D: row = i + mo, col = lane&15
  const int nn = lane & 15;
#pragma unroll
  for (int ct = 0; ct < 8; ++ct) {
    const int gcol = col0 + 16 * ct + nn;
    const float bv = bias[gcol];
    const int sec = gcol >> 10;              // 0=q 1=k 2=v
    const int cc = gcol & 1023;
    const int hh = cc >> 6, dd = cc & 63;
#pragma unroll
    for (int i = 0; i < 8; ++i) {
      const int grow = row0 + 16 * wave + mo + i;
      const int bb = grow >> 11, n = grow & 2047;
      const _Float16 val = (_Float16)(acc[ct][i] + bv);
      if (sec == 0)
        q_ws[(((size_t)(bb * HEADS + hh)) * SEQ + n) * HDIM + dd] = val;
      else if (sec == 1)
        kT_ws[(((size_t)(bb * HEADS + hh)) * HDIM + dd) * SEQ + n] = val;
      else
        v_ws[(((size_t)(bb * HEADS + hh)) * SEQ + n) * HDIM + dd] = val;
    }
  }
}

// ---------------------------------------------------------------------------
// Kernel 2: flash attention per (b, h, 64-row Q chunk); K/V double-buffered
// through async-LDS DMA.
// ---------------------------------------------------------------------------
#define A_STR 72   // 64 + 8 pad halves (144B rows)
#define S_STR 68   // 64 + 4 pad floats

__global__ void __launch_bounds__(256)
attn_kernel(const _Float16* __restrict__ q_ws, const _Float16* __restrict__ kT_ws,
            const _Float16* __restrict__ v_ws, _Float16* __restrict__ attn_out) {
  __shared__ __attribute__((aligned(16))) _Float16 Qs[64 * A_STR];
  __shared__ __attribute__((aligned(16))) _Float16 Kt[2][64 * A_STR];
  __shared__ __attribute__((aligned(16))) _Float16 Vs[2][64 * A_STR];
  __shared__ __attribute__((aligned(16))) float S[64 * S_STR];
  __shared__ float row_max[64], row_sum[64], row_scale[64];

  const int t = threadIdx.x, wave = t >> 5, lane = t & 31;
  const int wr = wave >> 1, wc = wave & 1;   // 4 row-strips x 2 col-halves
  const int qb = blockIdx.x * 64;
  const int head = blockIdx.y, bat = blockIdx.z;
  const size_t base = ((size_t)(bat * HEADS + head)) * SEQ * HDIM;

  // stage K^T[d][key] / V[key][d] chunk jb..jb+63 into buffer `buf`
  auto stage_kv = [&](int buf, int jb) {
#pragma unroll
    for (int i = 0; i < 2; ++i) {
      int ch = i * 256 + t;                  // 8 chunks per 64-half row
      int r = ch >> 3, c8 = (ch & 7) << 3;
      async_copy_b128(&Kt[buf][r * A_STR + c8],
                      kT_ws + base + (size_t)r * SEQ + jb + c8);
      async_copy_b128(&Vs[buf][r * A_STR + c8],
                      v_ws + base + (size_t)(jb + r) * HDIM + c8);
    }
  };

  // prologue: Q chunk + first K/V chunk
#pragma unroll
  for (int i = 0; i < 2; ++i) {
    int ch = i * 256 + t;
    int r = ch >> 3, c8 = (ch & 7) << 3;
    async_copy_b128(&Qs[r * A_STR + c8],
                    q_ws + base + (size_t)(qb + r) * HDIM + c8);
  }
  stage_kv(0, 0);
  if (t < 64) { row_max[t] = -1e30f; row_sum[t] = 0.0f; }
  wait_async();
  __syncthreads();

  const v16h qa0 = load_frag_a(Qs, 16 * wr, 0, A_STR, lane);
  const v16h qa1 = load_frag_a(Qs, 16 * wr, 32, A_STR, lane);

  v8f o0 = (v8f){}, o1 = (v8f){};
  const int mo = (lane >> 4) << 3, nn = lane & 15;

  for (int j = 0; j < SEQ / 64; ++j) {
    const int cur = j & 1;
    if (j + 1 < SEQ / 64) stage_kv(cur ^ 1, (j + 1) * 64);  // prefetch overlap

    // S(rows 16*wr.., keys 32*wc..+31) = Q @ K^T * 1/sqrt(D)
#pragma unroll
    for (int ct = 0; ct < 2; ++ct) {
      const int keyt = 16 * (2 * wc + ct);
      v8f s = (v8f){};
      s = wmma_f16(qa0, load_frag_b(Kt[cur], 0, keyt, A_STR, lane), s);
      s = wmma_f16(qa1, load_frag_b(Kt[cur], 32, keyt, A_STR, lane), s);
#pragma unroll
      for (int i = 0; i < 8; ++i)
        S[(16 * wr + mo + i) * S_STR + keyt + nn] = s[i] * 0.125f;
    }
    __syncthreads();

    if (t < 64) {                            // running row max + rescale factor
      float m = row_max[t], cmax = -1e30f;
      for (int c = 0; c < 64; ++c) cmax = fmaxf(cmax, S[t * S_STR + c]);
      float mnew = fmaxf(m, cmax);
      row_scale[t] = __expf(m - mnew);
      row_max[t] = mnew;
    }
    __syncthreads();

#pragma unroll
    for (int i = 0; i < 16; ++i) {           // P = exp(S - m) in place (f32)
      int lin = i * 256 + t;
      int r = lin >> 6, c = lin & 63;
      S[r * S_STR + c] = __expf(S[r * S_STR + c] - row_max[r]);
    }
    __syncthreads();

    if (t < 64) {                            // running row sum
      float s = 0.0f;
      for (int c = 0; c < 64; ++c) s += S[t * S_STR + c];
      row_sum[t] = row_sum[t] * row_scale[t] + s;
    }
#pragma unroll
    for (int i = 0; i < 8; ++i) {            // rescale O accumulators
      const float f = row_scale[16 * wr + mo + i];
      o0[i] *= f;
      o1[i] *= f;
    }
    __syncthreads();

    // O(rows 16*wr.., d 32*wc..+31) += P @ V
#pragma unroll
    for (int ks = 0; ks < 2; ++ks) {
      v16h pa = load_frag_a_f32(S, 16 * wr, 32 * ks, S_STR, lane);
      o0 = wmma_f16(pa, load_frag_b(Vs[cur], 32 * ks, 32 * wc + 0, A_STR, lane), o0);
      o1 = wmma_f16(pa, load_frag_b(Vs[cur], 32 * ks, 32 * wc + 16, A_STR, lane), o1);
    }
    wait_async();                            // prefetch for j+1 has landed
    __syncthreads();
  }

  // O /= l ; write f16 [B, N, H*D]
#pragma unroll
  for (int i = 0; i < 8; ++i) {
    const int r = 16 * wr + mo + i;
    const float inv = 1.0f / row_sum[r];
    const size_t off =
        ((size_t)(bat * SEQ + qb + r)) * EMB + head * HDIM + 32 * wc;
    attn_out[off + nn] = (_Float16)(o0[i] * inv);
    attn_out[off + 16 + nn] = (_Float16)(o1[i] * inv);
  }
}

// ---------------------------------------------------------------------------
// Kernel 3: out = attn(f16) @ wouth + b_out  (fp32 result)
// ---------------------------------------------------------------------------
__global__ void __launch_bounds__(256)
out_gemm_kernel(const _Float16* __restrict__ attn, const _Float16* __restrict__ w,
                const float* __restrict__ bias, float* __restrict__ out) {
  __shared__ __attribute__((aligned(16))) _Float16 As[2][128 * G_AS];
  __shared__ __attribute__((aligned(16))) _Float16 Bs[2][32 * G_BS];
  const int t = threadIdx.x;
  const int wave = t >> 5, lane = t & 31;
  const int row0 = blockIdx.y * 128;
  const int col0 = blockIdx.x * 128;

  v8f acc[8];
#pragma unroll
  for (int i = 0; i < 8; ++i) acc[i] = (v8f){};

  auto stage = [&](int buf, int kk) {
#pragma unroll
    for (int i = 0; i < 2; ++i) {
      int ch = i * 256 + t;
      int r = ch >> 2, c8 = (ch & 3) << 3;
      async_copy_b128(&As[buf][r * G_AS + c8],
                      attn + (size_t)(row0 + r) * EMB + kk + c8);
    }
#pragma unroll
    for (int i = 0; i < 2; ++i) {
      int ch = i * 256 + t;
      int r = ch >> 4, c8 = (ch & 15) << 3;
      async_copy_b128(&Bs[buf][r * G_BS + c8],
                      w + (size_t)(kk + r) * EMB + col0 + c8);
    }
  };

  stage(0, 0);
  wait_async();
  __syncthreads();

  for (int kk = 0; kk < EMB; kk += 32) {
    const int cur = (kk >> 5) & 1;
    if (kk + 32 < EMB) stage(cur ^ 1, kk + 32);
    v16h a = load_frag_a(As[cur], 16 * wave, 0, G_AS, lane);
#pragma unroll
    for (int ct = 0; ct < 8; ++ct) {
      v16h b = load_frag_b(Bs[cur], 0, 16 * ct, G_BS, lane);
      acc[ct] = wmma_f16(a, b, acc[ct]);
    }
    wait_async();
    __syncthreads();
  }

  const int mo = (lane >> 4) << 3;
  const int nn = lane & 15;
#pragma unroll
  for (int ct = 0; ct < 8; ++ct) {
    const int gcol = col0 + 16 * ct + nn;
    const float bv = bias[gcol];
#pragma unroll
    for (int i = 0; i < 8; ++i) {
      const int grow = row0 + 16 * wave + mo + i;
      out[(size_t)grow * EMB + gcol] = acc[ct][i] + bv;
    }
  }
}

// ---------------------------------------------------------------------------
extern "C" void kernel_launch(void* const* d_in, const int* in_sizes, int n_in,
                              void* d_out, int out_size, void* d_ws,
                              size_t ws_size, hipStream_t stream) {
  (void)in_sizes; (void)n_in; (void)out_size; (void)ws_size;
  const float* x     = (const float*)d_in[0];
  const float* w_qkv = (const float*)d_in[1];
  const float* b_qkv = (const float*)d_in[2];
  const float* w_out = (const float*)d_in[3];
  const float* b_out = (const float*)d_in[4];
  float* out = (float*)d_out;

  // Workspace carve (f16 halves).
  const size_t N_X  = (size_t)ROWS * EMB;          // 4,194,304
  const size_t N_WQ = (size_t)EMB * QKVC;          // 3,145,728
  const size_t N_WO = (size_t)EMB * EMB;           // 1,048,576
  const size_t SEC  = (size_t)BATCH * HEADS * SEQ * HDIM;  // 4,194,304
  _Float16* xh      = (_Float16*)d_ws;
  _Float16* wqkvh   = xh + N_X;
  _Float16* wouth   = wqkvh + N_WQ;
  _Float16* q_ws    = wouth + N_WO;
  _Float16* kT_ws   = q_ws + SEC;
  _Float16* v_ws    = kT_ws + SEC;
  _Float16* attn_ws = v_ws + SEC;

  convert_f16_kernel<<<(int)(N_X  / 2048), 256, 0, stream>>>(x, xh, (int)N_X);
  convert_f16_kernel<<<(int)(N_WQ / 2048), 256, 0, stream>>>(w_qkv, wqkvh, (int)N_WQ);
  convert_f16_kernel<<<(int)(N_WO / 2048), 256, 0, stream>>>(w_out, wouth, (int)N_WO);

  dim3 g1(QKVC / 128, ROWS / 128);           // 24 x 32
  qkv_gemm_kernel<<<g1, 256, 0, stream>>>(xh, wqkvh, b_qkv, q_ws, kT_ws, v_ws);

  dim3 g2(SEQ / 64, HEADS, BATCH);           // 32 x 16 x 2
  attn_kernel<<<g2, 256, 0, stream>>>(q_ws, kT_ws, v_ws, attn_ws);

  dim3 g3(EMB / 128, ROWS / 128);            // 8 x 32
  out_gemm_kernel<<<g3, 256, 0, stream>>>(attn_ws, wouth, b_out, out);
}
